// transformify_19902878449795
// MI455X (gfx1250) — matrix operand order repaired and checked by reading, not compiled, not run on hardware
//
#include <hip/hip_runtime.h>
#include <hip/hip_bf16.h>
#include <math.h>
#include <stdint.h>

// ---------------------------------------------------------------------------
// MI455X (gfx1250) transformer block: LN1 -> 8-head attention -> proj -> LN2
// -> GELU MLP with residuals.  All GEMMs run through one tiled bf16 WMMA
// kernel (v_wmma_f32_16x16x32_bf16, fp32 accumulate) with double-buffered
// LDS staging fed by global_load_async_to_lds_b128 (ASYNCcnt path).
// ---------------------------------------------------------------------------

typedef __attribute__((ext_vector_type(16))) __bf16 v16bf;
typedef __attribute__((ext_vector_type(8)))  __bf16 v8bf;
typedef __attribute__((ext_vector_type(8)))  float  v8f;

#define D_MODEL 512
#define N_HEAD  8
#define BATCH   4
#define SEQ     2048
#define ROWS    (BATCH * SEQ)   // 8192
#define FFN     (4 * D_MODEL)   // 2048

// epilogue mode bits
enum { M_BF16 = 1, M_TRANS = 2, M_BIAS = 4, M_GELU = 8, M_RES = 16 };

__device__ __forceinline__ float gelu_exact(float x) {
  return 0.5f * x * (1.0f + erff(x * 0.70710678118654752f));
}

__device__ __forceinline__ void async_load_16B(unsigned lds_off, const void* gaddr) {
  asm volatile("global_load_async_to_lds_b128 %0, %1, off"
               :: "v"(lds_off), "v"((unsigned long long)(uintptr_t)gaddr)
               : "memory");
}

__device__ __forceinline__ void wait_async0() {
#if __has_builtin(__builtin_amdgcn_s_wait_asynccnt)
  __builtin_amdgcn_s_wait_asynccnt(0);
#else
  asm volatile("s_wait_asynccnt 0x0" ::: "memory");
#endif
}

// ---------------- elementwise f32 -> bf16 ----------------------------------
__global__ void cvt_bf16_kernel(const float* __restrict__ in,
                                __bf16* __restrict__ out, int n) {
  int i = blockIdx.x * blockDim.x + threadIdx.x;
  if (i < n) out[i] = (__bf16)in[i];
}

// ---------------- layernorm over D=512, one block per row ------------------
__global__ __launch_bounds__(256)
void ln_kernel(const float* __restrict__ x, const float* __restrict__ g,
               const float* __restrict__ b, float* __restrict__ of32,
               __bf16* __restrict__ obf) {
  __shared__ float ssum[256];
  __shared__ float ssq[256];
  const int row = blockIdx.x;
  const int tid = threadIdx.x;
  const float* xr = x + (size_t)row * D_MODEL;
  float v0 = xr[tid];
  float v1 = xr[tid + 256];
  ssum[tid] = v0 + v1;
  ssq[tid]  = v0 * v0 + v1 * v1;
  __syncthreads();
  for (int o = 128; o > 0; o >>= 1) {
    if (tid < o) { ssum[tid] += ssum[tid + o]; ssq[tid] += ssq[tid + o]; }
    __syncthreads();
  }
  const float mean = ssum[0] * (1.0f / D_MODEL);
  const float var  = ssq[0] * (1.0f / D_MODEL) - mean * mean;
  const float rstd = rsqrtf(var + 1e-6f);
  float y0 = (v0 - mean) * rstd * g[tid] + b[tid];
  float y1 = (v1 - mean) * rstd * g[tid + 256] + b[tid + 256];
  of32[(size_t)row * D_MODEL + tid]       = y0;
  of32[(size_t)row * D_MODEL + tid + 256] = y1;
  obf[(size_t)row * D_MODEL + tid]        = (__bf16)y0;
  obf[(size_t)row * D_MODEL + tid + 256]  = (__bf16)y1;
}

// ---------------- softmax over SEQ, one block per row ----------------------
__global__ __launch_bounds__(256)
void softmax_kernel(const float* __restrict__ s, __bf16* __restrict__ p) {
  __shared__ float red[256];
  const int row = blockIdx.x;
  const int tid = threadIdx.x;
  const float* sr = s + (size_t)row * SEQ;
  __bf16* pr = p + (size_t)row * SEQ;
  float vals[8];
  float m = -3.0e38f;
#pragma unroll
  for (int i = 0; i < 8; ++i) { vals[i] = sr[tid + i * 256]; m = fmaxf(m, vals[i]); }
  red[tid] = m; __syncthreads();
  for (int o = 128; o > 0; o >>= 1) {
    if (tid < o) red[tid] = fmaxf(red[tid], red[tid + o]);
    __syncthreads();
  }
  m = red[0];
  __syncthreads();
  float sum = 0.0f;
#pragma unroll
  for (int i = 0; i < 8; ++i) { vals[i] = __expf(vals[i] - m); sum += vals[i]; }
  red[tid] = sum; __syncthreads();
  for (int o = 128; o > 0; o >>= 1) {
    if (tid < o) red[tid] += red[tid + o];
    __syncthreads();
  }
  const float inv = 1.0f / red[0];
#pragma unroll
  for (int i = 0; i < 8; ++i) pr[tid + i * 256] = (__bf16)(vals[i] * inv);
}

// ---------------- generic tiled bf16 WMMA GEMM -----------------------------
// C[M,N] = epilogue(alpha * A[M,K] @ B[K,N]).  Block tile 128x64, 8 waves,
// each wave a 32x32 quad of 16x16x32 bf16 WMMAs.  M%128==0, N%64==0, K%32==0.
// Double-buffered LDS; A tile staged via global_load_async_to_lds_b128.
#define LDA_S 40   // LDS strides (bf16 units), multiple of 8 => 16B aligned
#define LDB_S 40

template <unsigned MODE>
__global__ __launch_bounds__(256)
void gemm_bf16_kernel(const __bf16* __restrict__ A, int lda,
                      const __bf16* __restrict__ B, int ldb,
                      void* __restrict__ C, int ldc,
                      const float* __restrict__ bias,
                      const float* __restrict__ res, int ldr,
                      int K, float alpha) {
  __shared__ __align__(16) __bf16 lA[2][128 * LDA_S];  // [m][k]
  __shared__ __align__(16) __bf16 lB[2][64 * LDB_S];   // [n][k] (transposed)

  const int tid  = threadIdx.x;
  const int lane = tid & 31;
  const int wave = tid >> 5;
  const int wm   = wave >> 1;          // 0..3 -> M offset wm*32
  const int wn   = wave & 1;           // 0..1 -> N offset wn*32
  const int blockM = blockIdx.y * 128;
  const int blockN = blockIdx.x * 64;

  // A staging: 2 async 16B chunks / thread / tile (128 rows x 32 k)
  const int arow = tid >> 2;               // 0..63
  const int ac8  = (tid & 3) << 3;         // 0,8,16,24
  const __bf16* aG = A + (size_t)(blockM + arow) * lda + ac8;
  const size_t  aRowSkip = (size_t)64 * lda;          // chunk 2: rows 64..127
  const unsigned aL = (unsigned)(uintptr_t)&lA[0][arow * LDA_S + ac8];
  const unsigned aLSkip    = 64 * LDA_S * 2;          // bytes
  const unsigned aBufBytes = 128 * LDA_S * 2;

  // B staging: 1 coalesced 16B load + 8 transposed b16 stores / thread
  const int bk  = tid >> 3;                // 0..31
  const int bn8 = (tid & 7) << 3;          // 0..56
  const __bf16* bG = B + (size_t)bk * ldb + blockN + bn8;
  const size_t  bStep = (size_t)32 * ldb;

  v8f acc[2][2];
#pragma unroll
  for (int i = 0; i < 2; ++i)
#pragma unroll
    for (int j = 0; j < 2; ++j)
#pragma unroll
      for (int e = 0; e < 8; ++e) acc[i][j][e] = 0.0f;

  const int mlane = lane & 15;
  const int ka    = (lane >> 4) << 3;   // A frag k-offset: 0 or 8
  const int kb    = (lane >> 4) << 3;   // B frag k-offset: 0 or 8
  const int nTiles = K >> 5;

  // ---- prologue: stage tile 0 into buffer 0 ----
  {
    async_load_16B(aL, aG);
    async_load_16B(aL + aLSkip, aG + aRowSkip);
    v8bf breg = *(const v8bf*)bG;
#pragma unroll
    for (int e = 0; e < 8; ++e) lB[0][(bn8 + e) * LDB_S + bk] = breg[e];
  }
  wait_async0();
  __syncthreads();

  for (int t = 0; t < nTiles; ++t) {
    const int cur = t & 1;
    const int nxt = cur ^ 1;
    const bool hasNext = (t + 1 < nTiles);

    // kick off next tile: async A -> LDS, B -> register (held over compute)
    v8bf bregN;
    if (hasNext) {
      const __bf16* aGn = aG + (size_t)(t + 1) * 32;
      const unsigned aLn = aL + (unsigned)nxt * aBufBytes;
      async_load_16B(aLn, aGn);
      async_load_16B(aLn + aLSkip, aGn + aRowSkip);
      bregN = *(const v8bf*)(bG + (size_t)(t + 1) * bStep);
    }

    // ---- fragments from current buffer (wave32 WMMA layouts) ----
    v16bf af[2], bfq[2];
#pragma unroll
    for (int i = 0; i < 2; ++i) {
      const __bf16* ap = &lA[cur][(size_t)(wm * 32 + i * 16 + mlane) * LDA_S];
      v8bf lo = *(const v8bf*)(ap + ka);
      v8bf hi = *(const v8bf*)(ap + 16 + ka);
      af[i] = __builtin_shufflevector(lo, hi, 0,1,2,3,4,5,6,7,8,9,10,11,12,13,14,15);
    }
#pragma unroll
    for (int j = 0; j < 2; ++j) {
      const __bf16* bp = &lB[cur][(size_t)(wn * 32 + j * 16 + mlane) * LDB_S + kb];
      v8bf lo = *(const v8bf*)(bp);
      v8bf hi = *(const v8bf*)(bp + 16);
      bfq[j] = __builtin_shufflevector(lo, hi, 0,1,2,3,4,5,6,7,8,9,10,11,12,13,14,15);
    }
#pragma unroll
    for (int i = 0; i < 2; ++i)
#pragma unroll
      for (int j = 0; j < 2; ++j)
        acc[i][j] = __builtin_amdgcn_wmma_f32_16x16x32_bf16(
            false, af[i], false, bfq[j], (short)0, acc[i][j], false, false);

    // transpose-store next B tile while WMMAs drain
    if (hasNext) {
#pragma unroll
      for (int e = 0; e < 8; ++e) lB[nxt][(bn8 + e) * LDB_S + bk] = bregN[e];
    }

    wait_async0();
    __syncthreads();
  }

  // ---- epilogue ----
  const int mb = ((lane >> 4) & 1) * 8;
  const int nn = lane & 15;
#pragma unroll
  for (int i = 0; i < 2; ++i)
#pragma unroll
    for (int j = 0; j < 2; ++j)
#pragma unroll
      for (int r = 0; r < 8; ++r) {
        int m = blockM + wm * 32 + i * 16 + mb + r;
        int n = blockN + wn * 32 + j * 16 + nn;
        float v = acc[i][j][r] * alpha;
        if (MODE & M_BIAS) v += bias[n];
        if (MODE & M_GELU) v = gelu_exact(v);
        if (MODE & M_RES)  v += res[(size_t)m * ldr + n];
        size_t off = (MODE & M_TRANS) ? ((size_t)n * ldc + m)
                                      : ((size_t)m * ldc + n);
        if (MODE & M_BF16) ((__bf16*)C)[off] = (__bf16)v;
        else               ((float*)C)[off]  = v;
      }
}

template <unsigned MODE>
static void gemm(hipStream_t st, int M, int N, int K,
                 const __bf16* A, int lda, const __bf16* B, int ldb,
                 void* C, int ldc, const float* bias,
                 const float* res, int ldr, float alpha) {
  dim3 grid(N / 64, M / 128);
  gemm_bf16_kernel<MODE><<<grid, 256, 0, st>>>(A, lda, B, ldb, C, ldc,
                                               bias, res, ldr, K, alpha);
}

// ---------------------------------------------------------------------------
extern "C" void kernel_launch(void* const* d_in, const int* in_sizes, int n_in,
                              void* d_out, int out_size, void* d_ws, size_t ws_size,
                              hipStream_t stream) {
  const float* x   = (const float*)d_in[0];
  const float* Q   = (const float*)d_in[1];
  const float* Kw  = (const float*)d_in[2];
  const float* g1  = (const float*)d_in[3];
  const float* be1 = (const float*)d_in[4];
  const float* Wp  = (const float*)d_in[5];
  const float* bp  = (const float*)d_in[6];
  const float* g2  = (const float*)d_in[7];
  const float* be2 = (const float*)d_in[8];
  const float* W1  = (const float*)d_in[9];
  const float* b1  = (const float*)d_in[10];
  const float* W2  = (const float*)d_in[11];
  const float* b2  = (const float*)d_in[12];
  float* out = (float*)d_out;

  char* ws = (char*)d_ws;
  size_t off = 0;
  auto alloc = [&](size_t bytes) -> void* {
    void* p = ws + off;
    off += (bytes + 255) & ~(size_t)255;
    return p;
  };
  float*  xn_f32 = (float*)alloc((size_t)ROWS * D_MODEL * 4);
  __bf16* xn_bf  = (__bf16*)alloc((size_t)ROWS * D_MODEL * 2);
  __bf16* Qbf    = (__bf16*)alloc((size_t)N_HEAD * D_MODEL * D_MODEL * 2);
  __bf16* Kbf    = (__bf16*)alloc((size_t)N_HEAD * D_MODEL * D_MODEL * 2);
  __bf16* Wpbf   = (__bf16*)alloc((size_t)N_HEAD * D_MODEL * D_MODEL * 2);
  __bf16* W1bf   = (__bf16*)alloc((size_t)D_MODEL * FFN * 2);
  __bf16* W2bf   = (__bf16*)alloc((size_t)FFN * D_MODEL * 2);
  __bf16* qxh    = (__bf16*)alloc((size_t)ROWS * D_MODEL * 2);      // per-head Q proj
  __bf16* kxth   = (__bf16*)alloc((size_t)D_MODEL * ROWS * 2);      // per-head K proj, transposed
  float*  scores = (float*)alloc((size_t)SEQ * SEQ * 4);            // per (h,b)
  __bf16* probs  = (__bf16*)alloc((size_t)SEQ * SEQ * 2);
  __bf16* cat    = (__bf16*)alloc((size_t)ROWS * (N_HEAD * D_MODEL) * 2);
  float*  yraw   = (float*)alloc((size_t)ROWS * D_MODEL * 4);
  float*  yln    = (float*)alloc((size_t)ROWS * D_MODEL * 4);
  __bf16* ylnbf  = (__bf16*)alloc((size_t)ROWS * D_MODEL * 2);
  // h1 (8192x2048 bf16 = 33.5MB) reuses qxh..probs region (42MB, contiguous)
  __bf16* h1 = qxh;

  auto cvt = [&](const float* in, __bf16* o, int n) {
    cvt_bf16_kernel<<<(n + 255) / 256, 256, 0, stream>>>(in, o, n);
  };
  cvt(Q,  Qbf,  N_HEAD * D_MODEL * D_MODEL);
  cvt(Kw, Kbf,  N_HEAD * D_MODEL * D_MODEL);
  cvt(Wp, Wpbf, N_HEAD * D_MODEL * D_MODEL);
  cvt(W1, W1bf, D_MODEL * FFN);
  cvt(W2, W2bf, FFN * D_MODEL);

  // LN1
  ln_kernel<<<ROWS, 256, 0, stream>>>(x, g1, be1, xn_f32, xn_bf);

  const float scale = 0.044194173824159216f;  // 1/sqrt(512)
  for (int h = 0; h < N_HEAD; ++h) {
    // Qx_h = xn @ Q[h]          [8192,512]
    gemm<M_BF16>(stream, ROWS, D_MODEL, D_MODEL,
                 xn_bf, D_MODEL, Qbf + (size_t)h * D_MODEL * D_MODEL, D_MODEL,
                 qxh, D_MODEL, nullptr, nullptr, 0, 1.0f);
    // KxT_h = (xn @ K[h])^T     [512,8192]
    gemm<M_BF16 | M_TRANS>(stream, ROWS, D_MODEL, D_MODEL,
                 xn_bf, D_MODEL, Kbf + (size_t)h * D_MODEL * D_MODEL, D_MODEL,
                 kxth, ROWS, nullptr, nullptr, 0, 1.0f);
    for (int b = 0; b < BATCH; ++b) {
      // scores = Qx_h[b] @ KxT_h[:,b] * scale   [2048,2048] fp32
      gemm<0>(stream, SEQ, SEQ, D_MODEL,
              qxh + (size_t)b * SEQ * D_MODEL, D_MODEL,
              kxth + (size_t)b * SEQ, ROWS,
              scores, SEQ, nullptr, nullptr, 0, scale);
      softmax_kernel<<<SEQ, 256, 0, stream>>>(scores, probs);
      // cat[b,:,h*512:] = probs @ xn[b] + xn[b]
      gemm<M_BF16 | M_RES>(stream, SEQ, D_MODEL, SEQ,
              probs, SEQ,
              xn_bf + (size_t)b * SEQ * D_MODEL, D_MODEL,
              cat + (size_t)b * SEQ * (N_HEAD * D_MODEL) + (size_t)h * D_MODEL,
              N_HEAD * D_MODEL,
              nullptr, xn_f32 + (size_t)b * SEQ * D_MODEL, D_MODEL, 1.0f);
    }
  }

  // y = cat @ Wp + bp   [8192,512] fp32
  gemm<M_BIAS>(stream, ROWS, D_MODEL, N_HEAD * D_MODEL,
               cat, N_HEAD * D_MODEL, Wpbf, D_MODEL,
               yraw, D_MODEL, bp, nullptr, 0, 1.0f);
  // LN2
  ln_kernel<<<ROWS, 256, 0, stream>>>(yraw, g2, be2, yln, ylnbf);
  // h1 = gelu(yln @ W1 + b1)   [8192,2048] bf16
  gemm<M_BF16 | M_BIAS | M_GELU>(stream, ROWS, FFN, D_MODEL,
               ylnbf, D_MODEL, W1bf, FFN,
               h1, FFN, b1, nullptr, 0, 1.0f);
  // out = yln + gelu(h1 @ W2 + b2)   [8192,512] fp32
  gemm<M_BIAS | M_GELU | M_RES>(stream, ROWS, D_MODEL, FFN,
               h1, FFN, W2bf, D_MODEL,
               out, D_MODEL, b2, yln, D_MODEL, 1.0f);
}
